// GATClassifier_78649441124636
// MI455X (gfx1250) — matrix operand order repaired and checked by reading, not compiled, not run on hardware
//
#include <hip/hip_runtime.h>
#include <hip/hip_bf16.h>

// ---------------- problem constants ----------------
#define NNODES 200
#define NP     208          // 13 * 16 (tile-padded node count)
#define NTILE  13
#define KP     224          // 7 * 32 (K-padded for attn@v WMMA)
#define KQ     32           // K-padded head dim for q.k^T WMMA
#define HID    64
#define HEADS  4
#define HD     16
#define BATCH  32
#define WWIN   40
#define MWIN   (BATCH*WWIN) // 1280
#define EPS_LN 1e-5f

typedef __attribute__((ext_vector_type(16))) _Float16 v16h;
typedef __attribute__((ext_vector_type(8)))  float    v8f;
typedef _Float16 h16;

// ---------------- WMMA helpers (gfx1250, wave32) ----------------
// All fragment loads are UNCONDITIONAL and contiguous per lane (zero-padding
// is done physically in LDS at fill time) so they lower to ds_load_b128.

// A fragment: row-major A[rows][ld], tile (row0, k0..k0+31).
// Layout (this code's contiguous k order, not the ISA table order): lane = row%16, half = lane>>4;
// v16h[0..15] = K(half*16 + 0..15) (contiguous, matches wmma_load_B_cc).
static __device__ __forceinline__ v16h wmma_load_A(const h16* base, int ld,
                                                   int row0, int k0) {
  int lane = threadIdx.x & 31;
  int row  = row0 + (lane & 15);
  int half = lane >> 4;
  const h16* p0 = base + (long)row * ld + k0 + half * 16;       // 8 contiguous
  const h16* p1 = p0 + 8;                                       // 8 contiguous
  v16h a;
#pragma unroll
  for (int i = 0; i < 8; ++i) a[i] = p0[i];
#pragma unroll
  for (int i = 0; i < 8; ++i) a[8 + i] = p1[i];
  return a;
}

// B fragment from a column-contiguous store Bc[n][k] (K fastest):
// B(kk, nn) = Bc[nn * ldk + kk].  Per-lane: 16 contiguous f16.
// Lanes 0-15 carry K group k0..k0+15, lanes 16-31 carry k0+16..k0+31.
static __device__ __forceinline__ v16h wmma_load_B_cc(const h16* Bc, int ldk,
                                                      int k0, int n0) {
  int lane = threadIdx.x & 31;
  int col  = n0 + (lane & 15);
  const h16* p = Bc + (long)col * ldk + k0 + (lane >> 4) * 16;  // 16 contiguous
  v16h b;
#pragma unroll
  for (int j = 0; j < 16; ++j) b[j] = p[j];
  return b;
}

// C/D layout (ISA 7.12.2): lane l, vgpr r -> (row0 + r + (l>>4)*8, col0 + l&15)
static __device__ __forceinline__ v8f wmma_load_C_f32(const float* C, int ldc,
                                                      int row0, int col0) {
  int lane  = threadIdx.x & 31;
  int col   = col0 + (lane & 15);
  int rbase = row0 + ((lane >> 4) << 3);
  v8f c;
#pragma unroll
  for (int r = 0; r < 8; ++r) c[r] = C[(long)(rbase + r) * ldc + col];
  return c;
}

static __device__ __forceinline__ void wmma_store_C_f32(float* C, int ldc,
                                                        int row0, int col0, v8f c) {
  int lane  = threadIdx.x & 31;
  int col   = col0 + (lane & 15);
  int rbase = row0 + ((lane >> 4) << 3);
#pragma unroll
  for (int r = 0; r < 8; ++r) C[(long)(rbase + r) * ldc + col] = c[r];
}

static __device__ __forceinline__ v8f wmma_f16(v16h a, v16h b, v8f c) {
  // 8 args: (neg_a, A, neg_b, B, c_mod, C, reuse_a, reuse_b)
  return __builtin_amdgcn_wmma_f32_16x16x32_f16(false, a, false, b,
                                                (short)0, c, false, false);
}

// ---------------- graph normalization ----------------
__global__ void k_deg(const float* __restrict__ adj, float* __restrict__ dis) {
  int n = threadIdx.x;
  if (n < NNODES) {
    float d = 1.0f; // self loop
    for (int j = 0; j < NNODES; ++j) d += adj[n * NNODES + j];
    dis[n] = (d > 0.f) ? rsqrtf(d) : 0.f;
  }
}

// Additive attention-mask bias: (1 - adjn)*-1e9 (pad cols/rows -> -1e9)
__global__ void k_bias(const float* __restrict__ adj, const float* __restrict__ dis,
                       float* __restrict__ bias) {
  int idx = blockIdx.x * blockDim.x + threadIdx.x;
  if (idx >= NP * NP) return;
  int i = idx / NP, j = idx % NP;
  float b = -1e9f;
  if (i < NNODES && j < NNODES) {
    float a    = adj[i * NNODES + j] + (i == j ? 1.f : 0.f);
    float adjn = dis[i] * a * dis[j];
    b = (1.f - adjn) * -1e9f;
  }
  bias[idx] = b;
}

// ---------------- layer-1 qkv (rank-1, in_dim = 1) ----------------
__global__ void k_qkv1(const float* __restrict__ X,
                       const float* __restrict__ wq, const float* __restrict__ bq,
                       const float* __restrict__ wk, const float* __restrict__ bk,
                       const float* __restrict__ wv, const float* __restrict__ bv,
                       h16* __restrict__ Q, h16* __restrict__ K, h16* __restrict__ V) {
  long idx = (long)blockIdx.x * blockDim.x + threadIdx.x;
  if (idx >= (long)MWIN * NP * HID) return;
  int  c = (int)(idx % HID);
  long t = idx / HID;
  int  n = (int)(t % NP);
  int  m = (int)(t / NP);
  float q = 0.f, k = 0.f, v = 0.f;
  if (n < NNODES) {
    float x = X[(long)m * NNODES + n];
    q = x * wq[c] + bq[c];
    k = x * wk[c] + bk[c];
    v = x * wv[c] + bv[c];
  }
  Q[idx] = (h16)q; K[idx] = (h16)k; V[idx] = (h16)v;
}

// ---------------- fused per-(window,head) attention ----------------
// LDS: qh[NP*KQ] kh[NP*KQ] vT[HD*KP] f16 | sc[NP*NP] f32 | at[NP*KP] f16
// q,k stored [node][32] with cols 16..31 physically zero; v stored transposed
// [c][224] with nodes >= 200 zero -> all WMMA operand loads are contiguous.
__global__ void __launch_bounds__(256)
k_attn(const h16* __restrict__ Q, const h16* __restrict__ K,
       const h16* __restrict__ V, const float* __restrict__ bias,
       h16* __restrict__ AO) {
  extern __shared__ char smem[];
  h16*   qh = (h16*)smem;                   // NP*KQ
  h16*   kh = qh + NP * KQ;                 // NP*KQ
  h16*   vT = kh + NP * KQ;                 // HD*KP (transposed)
  float* sc = (float*)(vT + HD * KP);       // NP*NP
  h16*   at = (h16*)(sc + NP * NP);         // NP*KP

  int m    = blockIdx.x / HEADS;
  int head = blockIdx.x % HEADS;
  const h16* Qm = Q + (long)m * NP * HID + head * HD;
  const h16* Km = K + (long)m * NP * HID + head * HD;
  const h16* Vm = V + (long)m * NP * HID + head * HD;

  // fill q/k [node][32]: fold 1/sqrt(hd)=0.25 into q; zero pads
  for (int i = threadIdx.x; i < NP * KQ; i += 256) {
    int n = i >> 5, c = i & 31;
    bool live = (c < HD) && (n < NNODES);
    qh[i] = live ? (h16)(0.25f * (float)Qm[(long)n * HID + c]) : (h16)0.f;
    kh[i] = live ? Km[(long)n * HID + c] : (h16)0.f;
  }
  // fill vT [c][224]: zero pad nodes
  for (int i = threadIdx.x; i < HD * KP; i += 256) {
    int c = i / KP, n = i % KP;
    vT[i] = (n < NNODES) ? Vm[(long)n * HID + c] : (h16)0.f;
  }
  __syncthreads();

  int wave = threadIdx.x >> 5;

  // scores = (0.25*q) @ k^T + mask_bias   (bias rides in as WMMA C)
  for (int t = wave; t < NTILE * NTILE; t += 8) {
    int i = t / NTILE, j = t % NTILE;
    v16h a = wmma_load_A(qh, KQ, i * 16, 0);
    v16h b = wmma_load_B_cc(kh, KQ, 0, j * 16);   // B(kk,nn)=kh[nn][kk]
    v8f  c = wmma_load_C_f32(bias, NP, i * 16, j * 16);
    c = wmma_f16(a, b, c);
    wmma_store_C_f32(sc, NP, i * 16, j * 16, c);
  }
  __syncthreads();

  // row softmax (f32): pass1 max (strict-compare, no NaN canonicalize),
  // pass2 exp + sum with exp written back into sc, pass3 scale -> f16 attn
  if (threadIdx.x < NP) {
    int r = threadIdx.x;
    float* scr = sc + (long)r * NP;
    h16*   atr = at + (long)r * KP;
    float mx = -3.4e38f;
    for (int j = 0; j < NP; ++j) {
      float x = scr[j];
      mx = (x > mx) ? x : mx;
    }
    float s = 0.f;
    for (int j = 0; j < NP; ++j) {
      float e = __expf(scr[j] - mx);
      scr[j] = e;
      s += e;
    }
    float inv = 1.f / s;
    for (int j = 0; j < NP; ++j) atr[j] = (h16)(scr[j] * inv);
    for (int j = NP; j < KP; ++j) atr[j] = (h16)0.f;
  }
  __syncthreads();

  // out = attn @ v : 13 row tiles, 7 K-chunks of 32
  h16* AOm = AO + (long)m * NP * HID + head * HD;
  for (int t = wave; t < NTILE; t += 8) {
    v8f c = {};
#pragma unroll
    for (int k0 = 0; k0 < KP; k0 += 32) {
      v16h a = wmma_load_A(at, KP, t * 16, k0);
      v16h b = wmma_load_B_cc(vT, KP, k0, 0);     // B(kk,nn)=vT[nn][kk]
      c = wmma_f16(a, b, c);
    }
    int lane  = threadIdx.x & 31;
    int col   = lane & 15;
    int rbase = t * 16 + ((lane >> 4) << 3);
#pragma unroll
    for (int r = 0; r < 8; ++r)
      AOm[(long)(rbase + r) * HID + col] = (h16)c[r];
  }
}

// ---------------- layer-2 qkv: three [208,64]x[64,64] WMMA GEMMs ----------------
// Weights staged TRANSPOSED in LDS (wsh[n][k]) so B loads are contiguous.
__global__ void __launch_bounds__(256)
k_qkv2(const h16* __restrict__ Hin,
       const float* __restrict__ wq, const float* __restrict__ bq,
       const float* __restrict__ wk, const float* __restrict__ bk,
       const float* __restrict__ wv, const float* __restrict__ bv,
       h16* __restrict__ Q, h16* __restrict__ K, h16* __restrict__ V) {
  __shared__ h16 hin[NP * HID];
  __shared__ h16 wsh[HID * HID];   // transposed: wsh[n*HID + k] = w[k*HID + n]
  int m = blockIdx.x;
  const h16* Hm = Hin + (long)m * NP * HID;
  for (int i = threadIdx.x; i < NP * HID; i += 256) hin[i] = Hm[i];

  const float* Ws[3] = {wq, wk, wv};
  const float* Bs[3] = {bq, bk, bv};
  h16* Out[3] = {Q + (long)m * NP * HID, K + (long)m * NP * HID,
                 V + (long)m * NP * HID};
  int wave = threadIdx.x >> 5, lane = threadIdx.x & 31;

  for (int p = 0; p < 3; ++p) {
    __syncthreads();
    for (int i = threadIdx.x; i < HID * HID; i += 256) {
      int n = i >> 6, k = i & 63;
      wsh[i] = (h16)Ws[p][k * HID + n];
    }
    __syncthreads();
    for (int t = wave; t < NTILE * 4; t += 8) {
      int i = t >> 2, j = t & 3;
      v8f c = {};
#pragma unroll
      for (int k0 = 0; k0 < HID; k0 += 32) {
        v16h a = wmma_load_A(hin, HID, i * 16, k0);
        v16h b = wmma_load_B_cc(wsh, HID, k0, j * 16);
        c = wmma_f16(a, b, c);
      }
      int col   = j * 16 + (lane & 15);
      int rbase = i * 16 + ((lane >> 4) << 3);
      float bb  = Bs[p][col];
#pragma unroll
      for (int r = 0; r < 8; ++r)
        Out[p][(long)(rbase + r) * HID + col] = (h16)(c[r] + bb);
    }
  }
}

// ---------------- output projection + LayerNorm + ReLU (fused) ----------------
// LDS: ain[NP*HID] f16 | wsh[HID*HID] f16 (transposed) | ob[NP*HID] f32
__global__ void __launch_bounds__(256)
k_proj_ln(const h16* __restrict__ AO, const float* __restrict__ wo,
          const float* __restrict__ bo, const float* __restrict__ g,
          const float* __restrict__ be, h16* __restrict__ Hout16,
          float* __restrict__ HoutF, int write_f32) {
  extern __shared__ char smem[];
  h16*   ain = (h16*)smem;                 // NP*HID
  h16*   wsh = ain + NP * HID;             // HID*HID (transposed)
  float* ob  = (float*)(wsh + HID * HID);  // NP*HID

  int m = blockIdx.x;
  const h16* Am = AO + (long)m * NP * HID;
  for (int i = threadIdx.x; i < NP * HID; i += 256) ain[i] = Am[i];
  for (int i = threadIdx.x; i < HID * HID; i += 256) {
    int n = i >> 6, k = i & 63;
    wsh[i] = (h16)wo[k * HID + n];
  }
  __syncthreads();

  int wave = threadIdx.x >> 5, lane = threadIdx.x & 31;
  for (int t = wave; t < NTILE * 4; t += 8) {
    int i = t >> 2, j = t & 3;
    v8f c = {};
#pragma unroll
    for (int k0 = 0; k0 < HID; k0 += 32) {
      v16h a = wmma_load_A(ain, HID, i * 16, k0);
      v16h b = wmma_load_B_cc(wsh, HID, k0, j * 16);
      c = wmma_f16(a, b, c);
    }
    int col   = j * 16 + (lane & 15);
    int rbase = i * 16 + ((lane >> 4) << 3);
    float bb  = bo[col];
#pragma unroll
    for (int r = 0; r < 8; ++r)
      ob[(long)(rbase + r) * HID + col] = c[r] + bb;
  }
  __syncthreads();

  if (threadIdx.x < NP) {
    int n = threadIdx.x;
    h16* o16 = Hout16 + (long)(m * NP + n) * HID;
    if (n < NNODES) {
      float mu = 0.f;
      for (int c = 0; c < HID; ++c) mu += ob[n * HID + c];
      mu *= (1.f / HID);
      float var = 0.f;
      for (int c = 0; c < HID; ++c) {
        float d = ob[n * HID + c] - mu;
        var += d * d;
      }
      var *= (1.f / HID);
      float is = rsqrtf(var + EPS_LN);
      float* of = HoutF + (long)(m * NNODES + n) * HID;
      for (int c = 0; c < HID; ++c) {
        float y = fmaxf((ob[n * HID + c] - mu) * is * g[c] + be[c], 0.f);
        o16[c] = (h16)y;
        if (write_f32) of[c] = y;
      }
    } else {
      for (int c = 0; c < HID; ++c) o16[c] = (h16)0.f;  // pad rows -> 0
    }
  }
}

// ---------------- mean over windows ----------------
__global__ void k_mean(const float* __restrict__ H, float* __restrict__ Hm) {
  int idx = blockIdx.x * blockDim.x + threadIdx.x;
  if (idx >= BATCH * NNODES * HID) return;
  int c = idx % HID;
  int n = (idx / HID) % NNODES;
  int b = idx / (HID * NNODES);
  float s = 0.f;
  for (int w = 0; w < WWIN; ++w)
    s += H[(long)((b * WWIN + w) * NNODES + n) * HID + c];
  Hm[idx] = s * (1.f / WWIN);
}

// ---------------- attention readout + MLP head (tiny) ----------------
__global__ void __launch_bounds__(256)
k_head(const float* __restrict__ Hm, const float* __restrict__ wsr,
       const float* __restrict__ bsr, const float* __restrict__ wh1,
       const float* __restrict__ bh1, const float* __restrict__ gh,
       const float* __restrict__ beh, const float* __restrict__ wh2,
       const float* __restrict__ bh2, float* __restrict__ out) {
  __shared__ float h[NNODES * HID];
  __shared__ float s[NNODES];
  __shared__ float pooled[HID];
  __shared__ float t1[HID];
  __shared__ float zz[HID];
  __shared__ float red[4];
  int b = blockIdx.x;
  const float* Hb = Hm + (long)b * NNODES * HID;
  for (int i = threadIdx.x; i < NNODES * HID; i += 256) h[i] = Hb[i];
  __syncthreads();
  if (threadIdx.x < NNODES) {
    float v = bsr[0];
    for (int c = 0; c < HID; ++c) v += h[threadIdx.x * HID + c] * wsr[c];
    s[threadIdx.x] = v;
  }
  __syncthreads();
  if (threadIdx.x == 0) {
    float mx = -3.4e38f;
    for (int n = 0; n < NNODES; ++n) mx = (s[n] > mx) ? s[n] : mx;
    red[0] = mx;
  }
  __syncthreads();
  if (threadIdx.x < NNODES) s[threadIdx.x] = __expf(s[threadIdx.x] - red[0]);
  __syncthreads();
  if (threadIdx.x == 0) {
    float sm = 0.f;
    for (int n = 0; n < NNODES; ++n) sm += s[n];
    red[1] = 1.f / sm;
  }
  __syncthreads();
  if (threadIdx.x < HID) {
    float p = 0.f;
    for (int n = 0; n < NNODES; ++n) p += s[n] * h[n * HID + threadIdx.x];
    pooled[threadIdx.x] = p * red[1];
  }
  __syncthreads();
  if (threadIdx.x < HID) {
    float v = bh1[threadIdx.x];
    for (int k = 0; k < HID; ++k) v += pooled[k] * wh1[k * HID + threadIdx.x];
    t1[threadIdx.x] = v;
  }
  __syncthreads();
  if (threadIdx.x == 0) {
    float mu = 0.f;
    for (int c = 0; c < HID; ++c) mu += t1[c];
    mu *= (1.f / HID);
    float var = 0.f;
    for (int c = 0; c < HID; ++c) { float d = t1[c] - mu; var += d * d; }
    var *= (1.f / HID);
    red[2] = mu; red[3] = rsqrtf(var + EPS_LN);
  }
  __syncthreads();
  if (threadIdx.x < HID)
    zz[threadIdx.x] = fmaxf((t1[threadIdx.x] - red[2]) * red[3] * gh[threadIdx.x] +
                                beh[threadIdx.x], 0.f);
  __syncthreads();
  if (threadIdx.x < 2) {
    float v = bh2[threadIdx.x];
    for (int c = 0; c < HID; ++c) v += zz[c] * wh2[c * 2 + threadIdx.x];
    out[b * 2 + threadIdx.x] = v;
  }
}

// ---------------- host launcher ----------------
extern "C" void kernel_launch(void* const* d_in, const int* in_sizes, int n_in,
                              void* d_out, int out_size, void* d_ws, size_t ws_size,
                              hipStream_t stream) {
  (void)in_sizes; (void)n_in; (void)out_size; (void)ws_size;
  const float* X   = (const float*)d_in[0];
  const float* adj = (const float*)d_in[1];
  const float* wq1 = (const float*)d_in[2];  const float* bq1 = (const float*)d_in[3];
  const float* wk1 = (const float*)d_in[4];  const float* bk1 = (const float*)d_in[5];
  const float* wv1 = (const float*)d_in[6];  const float* bv1 = (const float*)d_in[7];
  const float* wo1 = (const float*)d_in[8];  const float* bo1 = (const float*)d_in[9];
  const float* g1  = (const float*)d_in[10]; const float* be1 = (const float*)d_in[11];
  const float* wq2 = (const float*)d_in[12]; const float* bq2 = (const float*)d_in[13];
  const float* wk2 = (const float*)d_in[14]; const float* bk2 = (const float*)d_in[15];
  const float* wv2 = (const float*)d_in[16]; const float* bv2 = (const float*)d_in[17];
  const float* wo2 = (const float*)d_in[18]; const float* bo2 = (const float*)d_in[19];
  const float* g2  = (const float*)d_in[20]; const float* be2 = (const float*)d_in[21];
  const float* wsr = (const float*)d_in[22]; const float* bsr = (const float*)d_in[23];
  const float* wh1 = (const float*)d_in[24]; const float* bh1 = (const float*)d_in[25];
  const float* gh  = (const float*)d_in[26]; const float* beh = (const float*)d_in[27];
  const float* wh2 = (const float*)d_in[28]; const float* bh2 = (const float*)d_in[29];
  float* out = (float*)d_out;

  char* ws = (char*)d_ws;
  size_t off = 0;
  float* dis  = (float*)(ws + off); off += 1024;
  float* bias = (float*)(ws + off); off += (size_t)NP * NP * 4;
  off = (off + 255) & ~(size_t)255;
  const size_t szQKV = (size_t)MWIN * NP * HID * sizeof(h16);  // 34 MB each
  size_t offQ = off;
  h16* Qg    = (h16*)(ws + off); off += szQKV;
  h16* Kg    = (h16*)(ws + off); off += szQKV;
  h16* Vg    = (h16*)(ws + off); off += szQKV;
  h16* AOg   = (h16*)(ws + off); off += szQKV;
  h16* Hin16 = (h16*)(ws + off); off += szQKV;
  float* Hmean = (float*)(ws + off); off += (size_t)BATCH * NNODES * HID * 4;
  // layer-2 f32 output aliases the (then-dead) Q+K regions: 65.5MB <= 68MB
  float* HoutF = (float*)(ws + offQ);

  const size_t smem_attn = (size_t)(NP * KQ + NP * KQ + HD * KP) * sizeof(h16)
                         + (size_t)NP * NP * 4 + (size_t)NP * KP * sizeof(h16); // 300032
  const size_t smem_proj = (size_t)(NP * HID + HID * HID) * sizeof(h16)
                         + (size_t)NP * HID * 4;                                // 88064

  k_deg<<<1, 256, 0, stream>>>(adj, dis);
  k_bias<<<(NP * NP + 255) / 256, 256, 0, stream>>>(adj, dis, bias);

  long nq = (long)MWIN * NP * HID;
  k_qkv1<<<(unsigned)((nq + 255) / 256), 256, 0, stream>>>(
      X, wq1, bq1, wk1, bk1, wv1, bv1, Qg, Kg, Vg);

  // ---- layer 1 ----
  k_attn<<<MWIN * HEADS, 256, smem_attn, stream>>>(Qg, Kg, Vg, bias, AOg);
  k_proj_ln<<<MWIN, 256, smem_proj, stream>>>(AOg, wo1, bo1, g1, be1,
                                              Hin16, HoutF, 0);
  // ---- layer 2 ----
  k_qkv2<<<MWIN, 256, 0, stream>>>(Hin16, wq2, bq2, wk2, bk2, wv2, bv2,
                                   Qg, Kg, Vg);
  k_attn<<<MWIN * HEADS, 256, smem_attn, stream>>>(Qg, Kg, Vg, bias, AOg);
  k_proj_ln<<<MWIN, 256, smem_proj, stream>>>(AOg, wo2, bo2, g2, be2,
                                              Hin16, HoutF, 1);

  // ---- readout ----
  k_mean<<<(BATCH * NNODES * HID + 255) / 256, 256, 0, stream>>>(HoutF, Hmean);
  k_head<<<BATCH, 256, 0, stream>>>(Hmean, wsr, bsr, wh1, bh1, gh, beh,
                                    wh2, bh2, out);
}
